// FPSTokenizer_45741401702756
// MI455X (gfx1250) — matrix operand order repaired and checked by reading, not compiled, not run on hardware
//
// CDNA5 / gfx1250 FPS tokenizer — round 3.
// - FPS + kNN on (x,y,z,t) only; heavy MLP computed ONLY on gathered kNN rows
//   (16384 rows instead of 65536).
// - GEMMs: v_wmma_f32_16x16x32_bf16, f32 accumulate. Weights pre-transposed
//   to bf16 [N x Kpad]; B frag loaded once per K-step, reused across 2 row
//   tiles (MT=32). Dynamic LDS 97KB/block (safe under CU-mode LDS caps).
// - Async LDS staging via global_load_async_to_lds_b128 + s_wait_asynccnt.
// - NEW: native __bf16 casts for f32->bf16 (RNE, fewer VALU than manual pack);
//   FPS centroid broadcast through an LDS point cache (no L2 round trip on
//   the 128-iteration serial chain).
// Workspace budget: ~9.2 MB.

#include <hip/hip_runtime.h>
#include <hip/hip_bf16.h>
#include <stdint.h>

#define B_EV 8
#define NPB  8192
#define FEATC 6
#define KTOK 128
#define TOKD 768
#define KNNB 16
#define LDAE 776       // bf16 elements per LDS activation row (768 + 8 pad)
#define MT   32        // rows per MLP block (2 x 16-row WMMA tiles)
#define MLP_SMEM (2 * MT * LDAE * 2)
#define FPS_SMEM ((4 * NPB + 32 + 32 + 8) * 4)

typedef __attribute__((ext_vector_type(16))) __bf16 v16bf;
typedef __attribute__((ext_vector_type(8)))  float  v8f;

struct Bits32B { uint4 lo; uint4 hi; };

static __device__ __forceinline__ v16bf frag_from(uint4 lo, uint4 hi) {
  Bits32B b; b.lo = lo; b.hi = hi;
  return __builtin_bit_cast(v16bf, b);
}

static __device__ __forceinline__ uint16_t f2bf(float f) {
  __bf16 h = (__bf16)f;                       // RNE, native bf16 lowering
  return __builtin_bit_cast(uint16_t, h);
}
static __device__ __forceinline__ float bf2f(uint16_t h) {
  return __uint_as_float((uint32_t)h << 16);
}

static __device__ __forceinline__ void wave_argmax(float& v, int& i) {
#pragma unroll
  for (int m = 16; m > 0; m >>= 1) {
    float ov = __shfl_xor(v, m, 32);
    int   oi = __shfl_xor(i, m, 32);
    if (ov > v || (ov == v && oi < i)) { v = ov; i = oi; }
  }
}
static __device__ __forceinline__ void wave_argmin(float& v, int& i) {
#pragma unroll
  for (int m = 16; m > 0; m >>= 1) {
    float ov = __shfl_xor(v, m, 32);
    int   oi = __shfl_xor(i, m, 32);
    if (ov < v || (ov == v && oi < i)) { v = ov; i = oi; }
  }
}

// Async global->LDS 16B copy (CDNA5 ASYNCcnt path). lds_off = LDS byte offset
// (low 32 bits of a generic LDS pointer), gaddr = 64-bit global address.
static __device__ __forceinline__ void async_copy_b128(uint32_t lds_off,
                                                       const void* gptr) {
  asm volatile("global_load_async_to_lds_b128 %0, %1, off"
               :: "v"(lds_off), "v"((unsigned long long)(uintptr_t)gptr)
               : "memory");
}
static __device__ __forceinline__ void wait_asynccnt0() {
  asm volatile("s_wait_asynccnt 0x0" ::: "memory");
}

// ---------------------------------------------------------------------------
// Weight prep: W [din x dout] f32 row-major -> WT [dout x kpad] bf16 (W^T)
// ---------------------------------------------------------------------------
__global__ void prep_weights(const float* __restrict__ W, uint16_t* __restrict__ WT,
                             int din, int dout, int kpad) {
  long total = (long)dout * kpad;
  for (long e = (long)blockIdx.x * blockDim.x + threadIdx.x; e < total;
       e += (long)gridDim.x * blockDim.x) {
    int n = (int)(e / kpad), k = (int)(e % kpad);
    float v = (k < din) ? W[(long)k * dout + n] : 0.0f;
    WT[e] = f2bf(v);
  }
}

// ---------------------------------------------------------------------------
// FPS: one block per event, 1024 threads, points in registers (8/thread) AND
// cached in LDS (SoA) so the per-iteration centroid broadcast is a uniform
// ds_load instead of an L2 round trip. 2 barriers per serial iteration.
// ---------------------------------------------------------------------------
__global__ __launch_bounds__(1024)
void fps_kernel(const float* __restrict__ coords, const float* __restrict__ times,
                float* __restrict__ cents /* [B][K][4] */) {
  extern __shared__ char dyn_smem[];
  float* lx = (float*)dyn_smem;
  float* ly = lx + NPB;
  float* lz = ly + NPB;
  float* lt = lz + NPB;
  float* wv = lt + NPB;            // [32]
  int*   wi = (int*)(wv + 32);     // [32]
  int*   slast = wi + 32;

  const int b = blockIdx.x, t = threadIdx.x;
  const int wid = t >> 5, lane = t & 31;
  const float INF = __int_as_float(0x7f800000);
  float px[8], py[8], pz[8], pw[8], mind[8];
#pragma unroll
  for (int i = 0; i < 8; ++i) {
    int li = t + i * 1024;
    long gi = (long)b * NPB + li;
    px[i] = coords[gi * 3 + 0];
    py[i] = coords[gi * 3 + 1];
    pz[i] = coords[gi * 3 + 2];
    pw[i] = times[gi];
    mind[i] = INF;
    lx[li] = px[i]; ly[li] = py[i]; lz[li] = pz[i]; lt[li] = pw[i];
  }

  // first = argmax over 4D norms (ties -> lowest index, matching jnp.argmax)
  float bv = -INF; int bi = 0;
#pragma unroll
  for (int i = 0; i < 8; ++i) {
    float n = px[i]*px[i] + py[i]*py[i] + pz[i]*pz[i] + pw[i]*pw[i];
    int li = t + i * 1024;
    if (n > bv || (n == bv && li < bi)) { bv = n; bi = li; }
  }
  wave_argmax(bv, bi);
  if (lane == 0) { wv[wid] = bv; wi[wid] = bi; }
  __syncthreads();
  if (wid == 0) {
    float v = wv[lane]; int ii = wi[lane];
    wave_argmax(v, ii);
    if (lane == 0) *slast = ii;
  }
  __syncthreads();
  int last = *slast;

  for (int k = 0; k < KTOK; ++k) {
    // uniform-address LDS reads -> broadcast, no bank conflicts
    const float cx = lx[last], cy = ly[last], cz = lz[last], ct = lt[last];
    if (t == 0) {
      float* cw = cents + ((long)b * KTOK + k) * 4;
      cw[0] = cx; cw[1] = cy; cw[2] = cz; cw[3] = ct;
    }
    bv = -INF; bi = 0;
#pragma unroll
    for (int i = 0; i < 8; ++i) {
      float dx = px[i]-cx, dy = py[i]-cy, dz = pz[i]-cz, dw = pw[i]-ct;
      float d = dx*dx + dy*dy + dz*dz + dw*dw;
      mind[i] = fminf(mind[i], d);
      int li = t + i * 1024;
      if (mind[i] > bv || (mind[i] == bv && li < bi)) { bv = mind[i]; bi = li; }
    }
    wave_argmax(bv, bi);
    if (lane == 0) { wv[wid] = bv; wi[wid] = bi; }
    __syncthreads();
    if (wid == 0) {
      float v = wv[lane]; int ii = wi[lane];
      wave_argmax(v, ii);
      if (lane == 0) *slast = ii;
    }
    __syncthreads();
    last = *slast;
  }
}

// ---------------------------------------------------------------------------
// kNN: one block per centroid (1024 blocks), 256 threads; dists in LDS,
// 16 serial argmin extractions (ties -> lowest index => same set as top_k).
// ---------------------------------------------------------------------------
__global__ __launch_bounds__(256)
void knn_kernel(const float* __restrict__ coords, const float* __restrict__ times,
                const float* __restrict__ cents, int* __restrict__ knn) {
  const int g = blockIdx.x, t = threadIdx.x;
  const int b = g >> 7;
  const int wid = t >> 5, lane = t & 31;
  const float INF = __int_as_float(0x7f800000);
  __shared__ float dist[NPB];
  __shared__ float rv[8];
  __shared__ int   ri[8];
  const float cx = cents[g*4+0], cy = cents[g*4+1], cz = cents[g*4+2], ct = cents[g*4+3];
#pragma unroll
  for (int j = 0; j < NPB / 256; ++j) {
    int li = t + j * 256;
    long gi = (long)b * NPB + li;
    float dx = coords[gi*3+0]-cx, dy = coords[gi*3+1]-cy, dz = coords[gi*3+2]-cz;
    float dw = times[gi] - ct;
    dist[li] = dx*dx + dy*dy + dz*dz + dw*dw;
  }
  __syncthreads();
  for (int r = 0; r < KNNB; ++r) {
    float bv = INF; int bi = NPB;
#pragma unroll 4
    for (int j = 0; j < NPB / 256; ++j) {
      int li = t + j * 256;
      float v = dist[li];
      if (v < bv || (v == bv && li < bi)) { bv = v; bi = li; }
    }
    wave_argmin(bv, bi);
    if (lane == 0) { rv[wid] = bv; ri[wid] = bi; }
    __syncthreads();
    if (wid == 0) {
      float v = (lane < 8) ? rv[lane] : INF;
      int  ii = (lane < 8) ? ri[lane] : 0x7fffffff;
      wave_argmin(v, ii);
      if (lane == 0) {
        knn[g * KNNB + r] = b * NPB + ii;   // global point row
        dist[ii] = INF;
      }
    }
    __syncthreads();
  }
}

// ---------------------------------------------------------------------------
// WMMA layer on a 32-row LDS tile: 8 waves split N; each wave loads the B
// fragment ONCE per K-step and applies it to BOTH 16-row tiles (2 WMMAs).
// A: LDS bf16 rows (stride LDAE). W: global bf16 [Nout x ldw] (= W^T).
// ---------------------------------------------------------------------------
static __device__ __forceinline__ void store8_bf16(uint16_t* out, int rowBase,
                                                   int n0, int r, int h,
                                                   v8f acc, float bs, bool relu) {
#pragma unroll
  for (int i = 0; i < 8; ++i) {
    float v = acc[i] + bs;
    if (relu) v = fmaxf(v, 0.f);
    out[(rowBase + i + h * 8) * LDAE + n0 + r] = f2bf(v);
  }
}

static __device__ __forceinline__ void layer_rt2(const uint16_t* __restrict__ in,
                                                 uint16_t* __restrict__ out,
                                                 const uint16_t* __restrict__ W,
                                                 int ldw, const float* __restrict__ bias,
                                                 int Kin, int Nout, bool relu) {
  const int w = threadIdx.x >> 5;
  const int lane = threadIdx.x & 31;
  const int h = lane >> 4, r = lane & 15;
  const int cw = Nout >> 3;  // columns per wave
  for (int n0 = w * cw; n0 < (w + 1) * cw; n0 += 16) {
    v8f acc0 = {0.f,0.f,0.f,0.f,0.f,0.f,0.f,0.f};
    v8f acc1 = {0.f,0.f,0.f,0.f,0.f,0.f,0.f,0.f};
    const uint16_t* wr  = W + (long)(n0 + r) * ldw;
    const uint16_t* ar0 = in + r * LDAE;
    const uint16_t* ar1 = in + (16 + r) * LDAE;
    for (int k0 = 0; k0 < Kin; k0 += 32) {
      uint4 blo = *(const uint4*)(wr + k0 + h * 8);
      uint4 bhi = *(const uint4*)(wr + k0 + 16 + h * 8);
      v16bf bf = frag_from(blo, bhi);
      uint4 alo0 = *(const uint4*)(ar0 + k0 + h * 8);
      uint4 ahi0 = *(const uint4*)(ar0 + k0 + 16 + h * 8);
      acc0 = __builtin_amdgcn_wmma_f32_16x16x32_bf16(
          false, frag_from(alo0, ahi0), false, bf, (short)0, acc0, false, false);
      uint4 alo1 = *(const uint4*)(ar1 + k0 + h * 8);
      uint4 ahi1 = *(const uint4*)(ar1 + k0 + 16 + h * 8);
      acc1 = __builtin_amdgcn_wmma_f32_16x16x32_bf16(
          false, frag_from(alo1, ahi1), false, bf, (short)0, acc1, false, false);
    }
    float bs = bias[n0 + r];
    store8_bf16(out, 0,  n0, r, h, acc0, bs, relu);
    store8_bf16(out, 16, n0, r, h, acc1, bs, relu);
  }
}

// ---------------------------------------------------------------------------
// Point MLP + max-pool: 1 block = 2 centroids = 32 gathered neighbor rows.
// 6->256->512->768->768, then per-centroid 16-row max-pool. Dynamic LDS.
// ---------------------------------------------------------------------------
__global__ __launch_bounds__(256)
void mlp_point_kernel(const int* __restrict__ knn, const float* __restrict__ features,
                      const uint16_t* __restrict__ wt1, const float* __restrict__ b1,
                      const uint16_t* __restrict__ wt2, const float* __restrict__ b2,
                      const uint16_t* __restrict__ wt3, const float* __restrict__ b3,
                      const uint16_t* __restrict__ wt4, const float* __restrict__ b4,
                      uint16_t* __restrict__ pooled) {
  extern __shared__ char dyn_smem[];
  uint16_t* bufA = (uint16_t*)dyn_smem;
  uint16_t* bufB = bufA + MT * LDAE;
  const int c0 = blockIdx.x * 2, t = threadIdx.x;
  if (t < MT) {
    long p = knn[(long)c0 * KNNB + t];  // rows 0..31 = [cent c0 | cent c0+1] x 16
#pragma unroll
    for (int c = 0; c < 32; ++c)
      bufA[t * LDAE + c] = (c < FEATC) ? f2bf(features[p * FEATC + c]) : (uint16_t)0;
  }
  __syncthreads();
  layer_rt2(bufA, bufB, wt1, 32,  b1, 32,  256, true);
  __syncthreads();
  layer_rt2(bufB, bufA, wt2, 256, b2, 256, 512, true);
  __syncthreads();
  layer_rt2(bufA, bufB, wt3, 512, b3, 512, 768, true);
  __syncthreads();
  layer_rt2(bufB, bufA, wt4, 768, b4, 768, 768, false);
  __syncthreads();
  // per-centroid max-pool over 16 neighbor rows (all valid for this input)
  for (int e = t; e < 2 * TOKD; e += 256) {
    int c = e / TOKD, col = e % TOKD;
    const uint16_t* base = bufA + c * 16 * LDAE;
    float m = bf2f(base[col]);
#pragma unroll
    for (int rr = 1; rr < 16; ++rr) m = fmaxf(m, bf2f(base[rr * LDAE + col]));
    pooled[(long)(c0 + c) * TOKD + col] = f2bf(m);
  }
}

// ---------------------------------------------------------------------------
// Neighborhood MLP 768->768(relu)->768 on 1024 pooled rows (32 blocks x 32).
// Pooled rows staged into LDS with async global->LDS copies (ASYNCcnt path).
// Final layer stores f32 straight to the tokens workspace.
// ---------------------------------------------------------------------------
__global__ __launch_bounds__(256)
void mlp_neigh_kernel(const uint16_t* __restrict__ pooled,
                      const uint16_t* __restrict__ wtn1, const float* __restrict__ bn1,
                      const uint16_t* __restrict__ wtn2, const float* __restrict__ bn2,
                      float* __restrict__ tokens) {
  extern __shared__ char dyn_smem[];
  uint16_t* bufA = (uint16_t*)dyn_smem;
  uint16_t* bufB = bufA + MT * LDAE;
  const int t = threadIdx.x;
  const int row0 = blockIdx.x * MT;
  // async-stage 32 rows x 768 bf16 (16B chunks; rows & LDS stride 16B aligned)
  for (int e = t; e < MT * (TOKD / 8); e += 256) {
    int row = e / (TOKD / 8), ch = e % (TOKD / 8);
    uint32_t lds_off = (uint32_t)(uintptr_t)(bufA + row * LDAE) + (uint32_t)ch * 16u;
    const void* src = (const uint8_t*)(pooled + (long)(row0 + row) * TOKD) + ch * 16;
    async_copy_b128(lds_off, src);
  }
  wait_asynccnt0();
  __syncthreads();
  layer_rt2(bufA, bufB, wtn1, 768, bn1, 768, 768, true);
  __syncthreads();
  // final layer: B-frag reused across both row tiles, f32 stores to global
  const int w = t >> 5, lane = t & 31, h = lane >> 4, r = lane & 15;
  for (int n0 = w * 96; n0 < (w + 1) * 96; n0 += 16) {
    v8f acc0 = {0.f,0.f,0.f,0.f,0.f,0.f,0.f,0.f};
    v8f acc1 = {0.f,0.f,0.f,0.f,0.f,0.f,0.f,0.f};
    const uint16_t* wr  = wtn2 + (long)(n0 + r) * 768;
    const uint16_t* ar0 = bufB + r * LDAE;
    const uint16_t* ar1 = bufB + (16 + r) * LDAE;
    for (int k0 = 0; k0 < 768; k0 += 32) {
      uint4 blo = *(const uint4*)(wr + k0 + h * 8);
      uint4 bhi = *(const uint4*)(wr + k0 + 16 + h * 8);
      v16bf bf = frag_from(blo, bhi);
      uint4 alo0 = *(const uint4*)(ar0 + k0 + h * 8);
      uint4 ahi0 = *(const uint4*)(ar0 + k0 + 16 + h * 8);
      acc0 = __builtin_amdgcn_wmma_f32_16x16x32_bf16(
          false, frag_from(alo0, ahi0), false, bf, (short)0, acc0, false, false);
      uint4 alo1 = *(const uint4*)(ar1 + k0 + h * 8);
      uint4 ahi1 = *(const uint4*)(ar1 + k0 + 16 + h * 8);
      acc1 = __builtin_amdgcn_wmma_f32_16x16x32_bf16(
          false, frag_from(alo1, ahi1), false, bf, (short)0, acc1, false, false);
    }
    float bs = bn2[n0 + r];
#pragma unroll
    for (int i = 0; i < 8; ++i)
      tokens[(long)(row0 + i + h * 8) * TOKD + n0 + r] = acc0[i] + bs;
#pragma unroll
    for (int i = 0; i < 8; ++i)
      tokens[(long)(row0 + 16 + i + h * 8) * TOKD + n0 + r] = acc1[i] + bs;
  }
}

// ---------------------------------------------------------------------------
// Sort-by-centroid-time (stable rank) + write tokens/cents/masks to d_out.
// ---------------------------------------------------------------------------
__global__ __launch_bounds__(256)
void finalize_kernel(const float* __restrict__ cents, const float* __restrict__ tokens,
                     float* __restrict__ out) {
  const long TOK_OFF = 0;
  const long CENT_OFF = (long)B_EV * KTOK * TOKD;          // 786432
  const long MASK_OFF = CENT_OFF + (long)B_EV * KTOK * 4;  // 790528
  const int b = blockIdx.x, t = threadIdx.x;
  __shared__ float tm[KTOK];
  __shared__ int rank[KTOK];
  if (t < KTOK) tm[t] = cents[((long)b * KTOK + t) * 4 + 3];
  __syncthreads();
  if (t < KTOK) {
    float ti = tm[t]; int rk = 0;
    for (int j = 0; j < KTOK; ++j) {
      float tj = tm[j];
      rk += (tj < ti || (tj == ti && j < t)) ? 1 : 0;
    }
    rank[t] = rk;
  }
  __syncthreads();
  for (int e = t; e < KTOK * TOKD; e += 256) {
    int i = e / TOKD, col = e % TOKD;
    out[TOK_OFF + ((long)(b * KTOK + rank[i])) * TOKD + col] =
        tokens[((long)(b * KTOK + i)) * TOKD + col];
  }
  for (int e = t; e < KTOK * 4; e += 256) {
    int i = e >> 2, c = e & 3;
    out[CENT_OFF + (long)(b * KTOK + rank[i]) * 4 + c] =
        cents[((long)b * KTOK + i) * 4 + c];
  }
  if (t < KTOK) out[MASK_OFF + (long)b * KTOK + t] = 1.0f;
}

// ---------------------------------------------------------------------------
// Workspace layout (bytes, all 256-aligned). Total ~9.2 MB.
// ---------------------------------------------------------------------------
static const size_t OFF_WT1   = 0;                       // 256*32*2   = 16384
static const size_t OFF_WT2   = 16384;                   // 512*256*2  = 262144
static const size_t OFF_WT3   = 278528;                  // 768*512*2  = 786432
static const size_t OFF_WT4   = 1064960;                 // 768*768*2  = 1179648
static const size_t OFF_WTN1  = 2244608;                 // 1179648
static const size_t OFF_WTN2  = 3424256;                 // 1179648
static const size_t OFF_CENTS = 4603904;                 // 1024*4*4   = 16384
static const size_t OFF_KNN   = 4620288;                 // 1024*16*4  = 65536
static const size_t OFF_POOL  = 4685824;                 // 1024*768*2 = 1572864
static const size_t OFF_TOKS  = 6258688;                 // 1024*768*4 = 3145728

extern "C" void kernel_launch(void* const* d_in, const int* in_sizes, int n_in,
                              void* d_out, int out_size, void* d_ws, size_t ws_size,
                              hipStream_t stream) {
  const float* coords   = (const float*)d_in[0];
  const float* features = (const float*)d_in[1];
  // d_in[2]: batch_ids — layout is fixed equal-sized & sorted, not needed.
  const float* times    = (const float*)d_in[3];
  const float* W1 = (const float*)d_in[4];   const float* b1 = (const float*)d_in[5];
  const float* W2 = (const float*)d_in[6];   const float* b2 = (const float*)d_in[7];
  const float* W3 = (const float*)d_in[8];   const float* b3 = (const float*)d_in[9];
  const float* W4 = (const float*)d_in[10];  const float* b4 = (const float*)d_in[11];
  const float* Wn1 = (const float*)d_in[12]; const float* bn1 = (const float*)d_in[13];
  const float* Wn2 = (const float*)d_in[14]; const float* bn2 = (const float*)d_in[15];

  char* ws = (char*)d_ws;
  uint16_t* wt1  = (uint16_t*)(ws + OFF_WT1);
  uint16_t* wt2  = (uint16_t*)(ws + OFF_WT2);
  uint16_t* wt3  = (uint16_t*)(ws + OFF_WT3);
  uint16_t* wt4  = (uint16_t*)(ws + OFF_WT4);
  uint16_t* wtn1 = (uint16_t*)(ws + OFF_WTN1);
  uint16_t* wtn2 = (uint16_t*)(ws + OFF_WTN2);
  float*    centsW  = (float*)(ws + OFF_CENTS);
  int*      knnW    = (int*)(ws + OFF_KNN);
  uint16_t* pooledW = (uint16_t*)(ws + OFF_POOL);
  float*    tokensW = (float*)(ws + OFF_TOKS);

  prep_weights<<<64,  256, 0, stream>>>(W1,  wt1,  6,   256, 32);
  prep_weights<<<256, 256, 0, stream>>>(W2,  wt2,  256, 512, 256);
  prep_weights<<<256, 256, 0, stream>>>(W3,  wt3,  512, 768, 512);
  prep_weights<<<256, 256, 0, stream>>>(W4,  wt4,  768, 768, 768);
  prep_weights<<<256, 256, 0, stream>>>(Wn1, wtn1, 768, 768, 768);
  prep_weights<<<256, 256, 0, stream>>>(Wn2, wtn2, 768, 768, 768);

  fps_kernel<<<B_EV, 1024, FPS_SMEM, stream>>>(coords, times, centsW);
  knn_kernel<<<B_EV * KTOK, 256, 0, stream>>>(coords, times, centsW, knnW);
  mlp_point_kernel<<<(B_EV * KTOK) / 2, 256, MLP_SMEM, stream>>>(
      knnW, features, wt1, b1, wt2, b2, wt3, b3, wt4, b4, pooledW);
  mlp_neigh_kernel<<<(B_EV * KTOK) / MT, 256, MLP_SMEM, stream>>>(
      pooledW, wtn1, bn1, wtn2, bn2, tokensW);
  finalize_kernel<<<B_EV, 256, 0, stream>>>(centsW, tokensW, (float*)d_out);
}
